// GatedConvParityNetwork_86277303042559
// MI455X (gfx1250) — matrix operand order repaired and checked by reading, not compiled, not run on hardware
//
#include <hip/hip_runtime.h>
#include <hip/hip_bf16.h>
#include <math.h>

typedef _Float16 v16h __attribute__((ext_vector_type(16)));
typedef float    v8f  __attribute__((ext_vector_type(8)));

#define N_NODES 10000
#define N_EDGES 160000
#define HID 100
#define NKB_H 4            // K = 128 (padded hidden) / 32

#define Y0C   0.28209479177387814f   // 1/(2 sqrt(pi))
#define Y1CC  0.4886025119029199f    // sqrt(3/(4 pi))
#define INV_SQRT3 0.5773502691896258f
#define INV_SQRT2 0.7071067811865476f
#define INV_SQRT_NNORM 0.25f         // 1/sqrt(E/N) = 1/4

// ---------------- per-wave LDS layout (bytes) ----------------
#define OFF_HA    0        // 2048 halves (16x128 h, A-operand order)
#define OFF_HB    4096
#define OFF_SJ    8192     // 16x16 f32
#define OFF_VDY   9216     // 16x16 f32
#define OFF_VJ    10240    // 16x16x3 f32
#define OFF_CR    13312    // 16x16x3 f32
#define OFF_YV    16384    // 16x3 f32
#define OFF_BAS   16640    // 16x3 f32
#define OFF_DST   16896    // 16 int
#define WAVE_LDS  17408
#define WAVES_PB  4

__device__ __forceinline__ float fast_sigmoid(float x){
  return __builtin_amdgcn_rcpf(1.f + __expf(-x));
}
__device__ __forceinline__ float fast_swish(float x){ return x * fast_sigmoid(x); }

// ---------------- weight repack: f32 [rows x cols] -> f16 tiled B-operand order ----------------
// block (nt,kb) = 512 halves: out[blk*512 + lane*16 + slot],
// K = kb*32 + 16*(lane>=16) + slot, N = nt*16 + (lane&15).
// Bias (if non-null) is packed as an extra K-row at K == rows (inside the zero pad);
// the matching A column is forced to 1.0 by the conv kernel.
__global__ void pack_kernel(const float* __restrict__ W, const float* __restrict__ bias,
                            int rows, int cols, int nkb,
                            _Float16* __restrict__ out, int total){
  int g = blockIdx.x*blockDim.x + threadIdx.x;
  if (g >= total) return;
  int slot = g & 15;
  int lane = (g >> 4) & 31;
  int blk  = g >> 9;
  int kb = blk % nkb;
  int nt = blk / nkb;
  int K = kb*32 + ((lane >= 16) ? 16 : 0) + slot;
  int Nc = nt*16 + (lane & 15);
  float v = 0.f;
  if (Nc < cols){
    if (K < rows)                 v = W[(size_t)K*cols + Nc];
    else if (bias && K == rows)   v = bias[Nc];
  }
  out[g] = (_Float16)v;
}

// ---------------- edge geometry: Yv (Y1C * rel/d), gaussian basis ----------------
__global__ void geom_kernel(const float* __restrict__ pos, const int* __restrict__ esrc,
                            const int* __restrict__ edst, float* __restrict__ geo){
  int e = blockIdx.x*blockDim.x + threadIdx.x;
  if (e >= N_EDGES) return;
  int s = esrc[e], d = edst[e];
  float rx = pos[(size_t)d*3+0]-pos[(size_t)s*3+0];
  float ry = pos[(size_t)d*3+1]-pos[(size_t)s*3+1];
  float rz = pos[(size_t)d*3+2]-pos[(size_t)s*3+2];
  float dd = sqrtf(rx*rx+ry*ry+rz*rz + 1e-12f);
  float inv = __builtin_amdgcn_rcpf(dd);
  float* g = geo + (size_t)e*6;
  g[0] = Y1CC*rx*inv; g[1] = Y1CC*ry*inv; g[2] = Y1CC*rz*inv;
  float u0 = dd*2.f, u1 = (dd-0.5f)*2.f, u2 = (dd-1.f)*2.f;  // width = 0.5
  g[3] = __expf(-u0*u0); g[4] = __expf(-u1*u1); g[5] = __expf(-u2*u2);
}

// ---------------- init node buffers from features ----------------
__global__ void init_kernel(const float* __restrict__ f, float* __restrict__ s, float* __restrict__ v){
  int idx = blockIdx.x*blockDim.x + threadIdx.x;
  if (idx >= N_NODES*16) return;
  int n = idx >> 4, m = idx & 15;
  s[idx] = f[(size_t)n*64 + m];
  v[(size_t)idx*3+0] = f[(size_t)n*64 + 16 + m*3 + 0];
  v[(size_t)idx*3+1] = f[(size_t)n*64 + 16 + m*3 + 1];
  v[(size_t)idx*3+2] = f[(size_t)n*64 + 16 + m*3 + 2];
}

// ---------------- gate / swish node update ----------------
__global__ void update_kernel(const float* __restrict__ sacc, const float* __restrict__ vacc,
                              float* __restrict__ s, float* __restrict__ v){
  int idx = blockIdx.x*blockDim.x + threadIdx.x;
  if (idx >= N_NODES*16) return;
  int n = idx >> 4, m = idx & 15;
  float sa = sacc[(size_t)n*32 + m];
  s[idx] = sa * fast_sigmoid(sa);
  float g = fast_sigmoid(sacc[(size_t)n*32 + 16 + m]);
  v[(size_t)idx*3+0] = vacc[(size_t)idx*3+0]*g;
  v[(size_t)idx*3+1] = vacc[(size_t)idx*3+1]*g;
  v[(size_t)idx*3+2] = vacc[(size_t)idx*3+2]*g;
}

// ---------------- finalize: out = [s | v.reshape(N,48)] ----------------
__global__ void finalize_kernel(const float* __restrict__ sacc, const float* __restrict__ vacc,
                                float* __restrict__ out){
  int idx = blockIdx.x*blockDim.x + threadIdx.x;
  if (idx >= N_NODES*16) return;
  int n = idx >> 4, m = idx & 15;
  out[(size_t)n*64 + m] = sacc[idx];                    // conv2 has ms_o = 16
  out[(size_t)n*64 + 16 + m*3 + 0] = vacc[(size_t)idx*3+0];
  out[(size_t)n*64 + 16 + m*3 + 1] = vacc[(size_t)idx*3+1];
  out[(size_t)n*64 + 16 + m*3 + 2] = vacc[(size_t)idx*3+2];
}

// ---------------- WMMA helpers ----------------
// store C-layout result (swish(x); col==HID forced to 1.0 = bias column) into LDS A-operand order
__device__ __forceinline__ void store_swish(_Float16* hdst, v8f acc, int col, int hi){
  int kb = col >> 5, k = col & 31;
  int laneA = ((k >> 4) & 1) * 16;
  int slot  = k & 15;
  #pragma unroll
  for (int r = 0; r < 8; ++r){
    float h = (col == HID) ? 1.0f : fast_swish(acc[r]);
    int M = r + hi*8;
    hdst[(kb*32 + laneA + M)*16 + slot] = (_Float16)h;
  }
}

__device__ __forceinline__ v8f wmma4(const v16h* a, v16h b0, v16h b1, v16h b2, v16h b3){
  v8f acc = {0.f,0.f,0.f,0.f,0.f,0.f,0.f,0.f};
  acc = __builtin_amdgcn_wmma_f32_16x16x32_f16(false, a[0], false, b0, (short)0, acc, false, false);
  acc = __builtin_amdgcn_wmma_f32_16x16x32_f16(false, a[1], false, b1, (short)0, acc, false, false);
  acc = __builtin_amdgcn_wmma_f32_16x16x32_f16(false, a[2], false, b2, (short)0, acc, false, false);
  acc = __builtin_amdgcn_wmma_f32_16x16x32_f16(false, a[3], false, b3, (short)0, acc, false, false);
  return acc;
}

__device__ __forceinline__ void hidden_layer(const _Float16* hsrc, _Float16* hdst,
                                             const v16h* __restrict__ wt,
                                             int lane, int hi, int nn){
  const v16h* hv = (const v16h*)hsrc;
  v16h a[NKB_H];
  #pragma unroll
  for (int kb=0;kb<NKB_H;kb++) a[kb] = hv[kb*32 + lane];
  for (int nt = 0; nt < 8; ++nt){
    const v16h* p = wt + (size_t)nt*128 + lane;
    v8f acc = wmma4(a, p[0], p[32], p[64], p[96]);
    store_swish(hdst, acc, nt*16 + nn, hi);
  }
}

// segment-dependent message accumulation for Wout o-tile t (wave-uniform branches)
__device__ __forceinline__ void consume_tile(int t, v8f acc, int soSub,
    int bnd1, int bnd2, int bnd3, int bnd4,
    const float* sjL, const float* vdL, const float* vjL, const float* crL, int hi,
    v8f& ms0, v8f& ms1, v8f& tsv, v8f& mvx, v8f& mvy, v8f& mvz)
{
  if (t < bnd2){                                 // ss / vs segments
    bool isVS = (t >= bnd1);
    int tr = isVS ? (t - bnd1) : t;
    int i   = (soSub == 2) ? (tr >> 1) : tr;
    int sub = (soSub == 2) ? (tr & 1) : 0;
    const float* src = isVS ? vdL : sjL;
    float scale = isVS ? 1.f : Y0C;
    if (sub == 0){
      #pragma unroll
      for (int r=0;r<8;r++){ float sv = src[(r+hi*8)*16+i]; ms0[r] += scale*sv*acc[r]; }
    } else {
      #pragma unroll
      for (int r=0;r<8;r++){ float sv = src[(r+hi*8)*16+i]; ms1[r] += scale*sv*acc[r]; }
    }
  } else if (t < bnd3){                          // sv: (w_sv . sj), x Yv at the end
    int i = t - bnd2;
    #pragma unroll
    for (int r=0;r<8;r++){ float sv = sjL[(r+hi*8)*16+i]; tsv[r] += sv*acc[r]; }
  } else if (t < bnd4){                          // vv0: Y0 * w_vv0 . vj
    int i = t - bnd3;
    #pragma unroll
    for (int r=0;r<8;r++){
      int M = r+hi*8;
      mvx[r] += Y0C*vjL[M*48+i*3+0]*acc[r];
      mvy[r] += Y0C*vjL[M*48+i*3+1]*acc[r];
      mvz[r] += Y0C*vjL[M*48+i*3+2]*acc[r];
    }
  } else {                                       // vv1: w_vv1 . cross(vj,Yv)/sqrt2
    int i = t - bnd4;
    #pragma unroll
    for (int r=0;r<8;r++){
      int M = r+hi*8;
      mvx[r] += crL[M*48+i*3+0]*acc[r];
      mvy[r] += crL[M*48+i*3+1]*acc[r];
      mvz[r] += crL[M*48+i*3+2]*acc[r];
    }
  }
}

// ---------------- fused conv: radial MLP (WMMA) + messages + atomic scatter ----------------
__global__ __launch_bounds__(128) void conv_kernel(
    const float* __restrict__ s_in, const float* __restrict__ v_in,
    const int* __restrict__ esrc, const int* __restrict__ edst,
    const float* __restrict__ geo,
    const v16h* __restrict__ w0t,
    const v16h* __restrict__ w1t,
    const v16h* __restrict__ w2t,
    const v16h* __restrict__ wot,
    float* __restrict__ s_acc, float* __restrict__ v_acc,
    int ms_o)
{
  extern __shared__ char smem[];
  const int lane = threadIdx.x & 31;
  const int wv   = threadIdx.x >> 5;
  char* wb = smem + (size_t)wv * WAVE_LDS;
  _Float16* hA = (_Float16*)(wb + OFF_HA);
  _Float16* hB = (_Float16*)(wb + OFF_HB);
  float* sjL = (float*)(wb + OFF_SJ);
  float* vdL = (float*)(wb + OFF_VDY);
  float* vjL = (float*)(wb + OFF_VJ);
  float* crL = (float*)(wb + OFF_CR);
  float* yvL = (float*)(wb + OFF_YV);
  float* baL = (float*)(wb + OFF_BAS);
  int*   dsL = (int*)(wb + OFF_DST);

  const int eBase = (blockIdx.x * WAVES_PB + wv) * 16;

  // ---- stage 16 edges: indices, geometry, gathered s/v rows ----
  if (lane < 16){
    int e = eBase + lane;
    int si = esrc[e];
    dsL[lane] = edst[e];
    const float* g = geo + (size_t)e*6;
    yvL[lane*3+0]=g[0]; yvL[lane*3+1]=g[1]; yvL[lane*3+2]=g[2];
    baL[lane*3+0]=g[3]; baL[lane*3+1]=g[4]; baL[lane*3+2]=g[5];
    const float4* sp = (const float4*)(s_in + (size_t)si*16);
    float4* sd = (float4*)(sjL + lane*16);
    #pragma unroll
    for (int k=0;k<4;k++) sd[k]=sp[k];
  } else {
    int e = eBase + (lane-16);
    int si = esrc[e];
    const float4* vp = (const float4*)(v_in + (size_t)si*48);
    float4* vd = (float4*)(vjL + (lane-16)*48);
    #pragma unroll
    for (int k=0;k<12;k++) vd[k]=vp[k];
  }
  __syncthreads();
  if (lane < 16){
    float yx=yvL[lane*3], yy=yvL[lane*3+1], yz=yvL[lane*3+2];
    #pragma unroll
    for (int i=0;i<16;i++){
      float vx=vjL[lane*48+i*3], vy=vjL[lane*48+i*3+1], vz=vjL[lane*48+i*3+2];
      vdL[lane*16+i]      = (vx*yx+vy*yy+vz*yz)*INV_SQRT3;
      crL[lane*48+i*3+0]  = (vy*yz - vz*yy)*INV_SQRT2;
      crL[lane*48+i*3+1]  = (vz*yx - vx*yz)*INV_SQRT2;
      crL[lane*48+i*3+2]  = (vx*yy - vy*yx)*INV_SQRT2;
    }
  }
  __syncthreads();

  const int hi = lane >> 4;
  const int nn = lane & 15;

  // ---- layer 1: basis(16x32, K=3 slot = 1.0 bias column) @ W0(+bias row) -> h1 ----
  v16h a0;
  #pragma unroll
  for (int i=0;i<16;i++) a0[i] = (_Float16)0.f;
  if (lane < 16){
    a0[0] = (_Float16)baL[lane*3+0];
    a0[1] = (_Float16)baL[lane*3+1];
    a0[2] = (_Float16)baL[lane*3+2];
    a0[3] = (_Float16)1.0f;                    // bias column (K=3)
  }
  for (int nt = 0; nt < 8; ++nt){
    v8f acc = {0.f,0.f,0.f,0.f,0.f,0.f,0.f,0.f};
    v16h b = w0t[nt*32 + lane];                // nkb = 1 for W0
    acc = __builtin_amdgcn_wmma_f32_16x16x32_f16(false, a0, false, b, (short)0, acc, false, false);
    store_swish(hA, acc, nt*16 + nn, hi);
  }
  // ---- layers 2,3 (bias via K=100 row, A col 100 == 1.0) ----
  hidden_layer(hA, hB, w1t, lane, hi, nn);
  hidden_layer(hB, hA, w2t, lane, hi, nn);

  // ---- fused Wout GEMM + equivariant message contraction (software pipelined) ----
  const v16h* hv = (const v16h*)hA;
  v16h a[NKB_H];
  #pragma unroll
  for (int kb=0;kb<NKB_H;kb++) a[kb] = hv[kb*32 + lane];

  v8f z = {0.f,0.f,0.f,0.f,0.f,0.f,0.f,0.f};
  v8f ms0 = z, ms1 = z, tsv = z, mvx = z, mvy = z, mvz = z;
  const int soSub = ms_o >> 4;   // 2 for conv0/1, 1 for conv2 (wave-uniform)
  const int bnd1 = 16*soSub, bnd2 = 32*soSub, bnd3 = bnd2+16, bnd4 = bnd3+16;
  const int nwT  = bnd4 + 16;    // 112 or 80 (even)

  v16h c0,c1,c2,c3, d0,d1,d2,d3;
  {
    const v16h* p = wot + lane;
    c0 = p[0]; c1 = p[32]; c2 = p[64]; c3 = p[96];
  }
  for (int t = 0; t < nwT; t += 2){
    {
      const v16h* p = wot + (size_t)(t+1)*128 + lane;
      d0 = p[0]; d1 = p[32]; d2 = p[64]; d3 = p[96];
    }
    v8f acc = wmma4(a, c0, c1, c2, c3);
    consume_tile(t, acc, soSub, bnd1, bnd2, bnd3, bnd4, sjL, vdL, vjL, crL, hi,
                 ms0, ms1, tsv, mvx, mvy, mvz);
    if (t + 2 < nwT){
      const v16h* p = wot + (size_t)(t+2)*128 + lane;
      c0 = p[0]; c1 = p[32]; c2 = p[64]; c3 = p[96];
    }
    v8f acc2 = wmma4(a, d0, d1, d2, d3);
    consume_tile(t+1, acc2, soSub, bnd1, bnd2, bnd3, bnd4, sjL, vdL, vjL, crL, hi,
                 ms0, ms1, tsv, mvx, mvy, mvz);
  }

  // ---- scatter with float atomics (segment_sum / sqrt(N_NORM)) ----
  #pragma unroll
  for (int r=0;r<8;r++){
    int M = r + hi*8;
    int dv = dsL[M];
    float yx=yvL[M*3], yy=yvL[M*3+1], yz=yvL[M*3+2];
    atomicAdd(&s_acc[(size_t)dv*ms_o + nn], ms0[r]*INV_SQRT_NNORM);
    if (soSub == 2)
      atomicAdd(&s_acc[(size_t)dv*ms_o + 16 + nn], ms1[r]*INV_SQRT_NNORM);
    float ox = (mvx[r] + tsv[r]*yx)*INV_SQRT_NNORM;
    float oy = (mvy[r] + tsv[r]*yy)*INV_SQRT_NNORM;
    float oz = (mvz[r] + tsv[r]*yz)*INV_SQRT_NNORM;
    size_t vb = ((size_t)dv*16 + nn)*3;
    atomicAdd(&v_acc[vb+0], ox);
    atomicAdd(&v_acc[vb+1], oy);
    atomicAdd(&v_acc[vb+2], oz);
  }
}

// ---------------- host orchestration ----------------
extern "C" void kernel_launch(void* const* d_in, const int* in_sizes, int n_in,
                              void* d_out, int out_size, void* d_ws, size_t ws_size,
                              hipStream_t stream)
{
  (void)in_sizes; (void)n_in; (void)out_size; (void)ws_size;
  const float* features = (const float*)d_in[0];
  const float* pos      = (const float*)d_in[1];
  const int*   esrc     = (const int*)d_in[2];
  const int*   edst     = (const int*)d_in[3];
  const float* P[3][7];
  for (int c=0;c<3;c++) for (int k=0;k<7;k++) P[c][k] = (const float*)d_in[4 + c*7 + k];

  char* ws = (char*)d_ws;
  constexpr size_t SZ_GEO  = (size_t)N_EDGES*6*4;
  constexpr size_t SZ_S    = (size_t)N_NODES*16*4;
  constexpr size_t SZ_V    = (size_t)N_NODES*48*4;
  constexpr size_t SZ_SACC = (size_t)N_NODES*32*4;
  constexpr size_t OFF_GEO  = 0;
  constexpr size_t OFF_SA   = OFF_GEO + SZ_GEO;
  constexpr size_t OFF_VA   = OFF_SA + SZ_S;
  constexpr size_t OFF_SB   = OFF_VA + SZ_V;
  constexpr size_t OFF_VB   = OFF_SB + SZ_S;
  constexpr size_t OFF_SACC = OFF_VB + SZ_V;
  constexpr size_t OFF_VACC = OFF_SACC + SZ_SACC;
  constexpr size_t OFF_WP   = OFF_VACC + SZ_V;

  float* geo  = (float*)(ws + OFF_GEO);
  float* sA   = (float*)(ws + OFF_SA);
  float* vA   = (float*)(ws + OFF_VA);
  float* sB   = (float*)(ws + OFF_SB);
  float* vB   = (float*)(ws + OFF_VB);
  float* sacc = (float*)(ws + OFF_SACC);
  float* vacc = (float*)(ws + OFF_VACC);

  _Float16* wp = (_Float16*)(ws + OFF_WP);
  _Float16 *w0p[3], *w1p[3], *w2p[3], *wop[3];
  size_t cur = 0;
  for (int c=0;c<3;c++){
    size_t hwout = (c<2) ? (size_t)112*4*512 : (size_t)80*4*512;
    w0p[c] = wp + cur; cur += (size_t)8*1*512;
    w1p[c] = wp + cur; cur += (size_t)8*4*512;
    w2p[c] = wp + cur; cur += (size_t)8*4*512;
    wop[c] = wp + cur; cur += hwout;
  }

  auto packL = [&](const float* W, const float* bias, int rows, int cols, _Float16* outp){
    int ntN = (cols + 15)/16;
    int nkb = (rows + 31)/32;
    int total = ntN*nkb*512;
    pack_kernel<<<(total+255)/256, 256, 0, stream>>>(W, bias, rows, cols, nkb, outp, total);
  };
  for (int c=0;c<3;c++){
    int nw = (c<2) ? 1792 : 1280;
    packL(P[c][0], P[c][1], 3,   HID, w0p[c]);
    packL(P[c][2], P[c][3], HID, HID, w1p[c]);
    packL(P[c][4], P[c][5], HID, HID, w2p[c]);
    packL(P[c][6], nullptr, HID, nw,  wop[c]);
  }

  init_kernel<<<(N_NODES*16+255)/256, 256, 0, stream>>>(features, sA, vA);
  geom_kernel<<<(N_EDGES+255)/256, 256, 0, stream>>>(pos, esrc, edst, geo);

  const size_t shmem = (size_t)WAVES_PB * WAVE_LDS;
  for (int li=0; li<3; ++li){
    int ms_o = (li<2) ? 32 : 16;
    hipMemsetAsync(sacc, 0, (size_t)N_NODES*ms_o*4, stream);
    hipMemsetAsync(vacc, 0, SZ_V, stream);
    const float* si = (li==1) ? sB : sA;
    const float* vi = (li==1) ? vB : vA;
    conv_kernel<<<N_EDGES/(WAVES_PB*16), 128, shmem, stream>>>(
        si, vi, esrc, edst, geo,
        (const v16h*)w0p[li],
        (const v16h*)w1p[li],
        (const v16h*)w2p[li],
        (const v16h*)wop[li],
        sacc, vacc, ms_o);
    if (li==0)      update_kernel<<<(N_NODES*16+255)/256, 256, 0, stream>>>(sacc, vacc, sB, vB);
    else if (li==1) update_kernel<<<(N_NODES*16+255)/256, 256, 0, stream>>>(sacc, vacc, sA, vA);
    else            finalize_kernel<<<(N_NODES*16+255)/256, 256, 0, stream>>>(sacc, vacc, (float*)d_out);
  }
}
